// DecoderRNN_62921270886429
// MI455X (gfx1250) — matrix operand order repaired and checked by reading, not compiled, not run on hardware
//
#include <hip/hip_runtime.h>
#include <hip/hip_bf16.h>

#define N_  64
#define E_  512
#define H_  512
#define L_  128
#define V_  32000

typedef __attribute__((ext_vector_type(16))) __bf16 v16bf;
typedef __attribute__((ext_vector_type(8)))  float  v8f;

// ---- helpers ---------------------------------------------------------------

__device__ __forceinline__ float sigmoidf(float x) {
  return 1.0f / (1.0f + __expf(-x));
}

__device__ __forceinline__ v8f wmma_bf16(v16bf a, v16bf b, v8f c) {
  // D = A(16x32 bf16) x B(32x16 bf16) + C(16x16 f32)
  return __builtin_amdgcn_wmma_f32_16x16x32_bf16(
      false, a, false, b, (short)0, c, false, false);
}

// A fragment (16x32, bf16) from f32 row-major matrix, one row pointer per lane
// lanes 0-15: row = lane, K = {k0..k0+7, k0+16..k0+23}
// lanes 16-31: row = lane-16, K = {k0+8..k0+15, k0+24..k0+31}
__device__ __forceinline__ v16bf load_a_frag_rowptr(const float* rowp, int k0) {
  const int lane = threadIdx.x & 31;
  const float* p = rowp + k0 + ((lane >> 4) << 3);
  v16bf u;
#pragma unroll
  for (int j = 0; j < 8; ++j) u[j] = (__bf16)p[j];
#pragma unroll
  for (int j = 0; j < 8; ++j) u[8 + j] = (__bf16)p[16 + j];
  return u;
}

__device__ __forceinline__ v16bf load_a_frag_f32(const float* A, int lda,
                                                 int row0, int k0) {
  const int lane = threadIdx.x & 31;
  return load_a_frag_rowptr(A + (size_t)(row0 + (lane & 15)) * lda, k0);
}

// A fragment from a bf16 row-major matrix (no conversion needed)
__device__ __forceinline__ v16bf load_a_frag_bf(const __bf16* A, int lda,
                                                int row0, int k0) {
  const int lane = threadIdx.x & 31;
  const __bf16* p =
      A + (size_t)(row0 + (lane & 15)) * lda + k0 + ((lane >> 4) << 3);
  v16bf u;
#pragma unroll
  for (int j = 0; j < 8; ++j) u[j] = p[j];
#pragma unroll
  for (int j = 0; j < 8; ++j) u[8 + j] = p[16 + j];
  return u;
}

// B fragment (32x16) taken TRANSPOSED from weight matrix W (Nout x K, f32
// row-major), i.e. B[k][n] = W[n0+n][k].  lane%16 = column, K = {k0+8h..+7, k0+16+8h..+7}, h = lane/16.
__device__ __forceinline__ v16bf load_b_fragT(const float* W, int ldw,
                                              int n0, int k0) {
  const int lane = threadIdx.x & 31;
  const float* p = W + (size_t)(n0 + (lane & 15)) * ldw + k0 + ((lane >> 4) << 3);
  v16bf u;
#pragma unroll
  for (int j = 0; j < 8; ++j) u[j] = (__bf16)p[j];
#pragma unroll
  for (int j = 0; j < 8; ++j) u[8 + j] = (__bf16)p[16 + j];
  return u;
}

// ---- kernels ---------------------------------------------------------------

__global__ __launch_bounds__(256) void k_zero(float* p, int n) {
  int i = blockIdx.x * blockDim.x + threadIdx.x;
  if (i < n) p[i] = 0.0f;
}

// Fused: embedding gather + LSTM gate GEMMs + cell update.
// grid = (H/16, N/16), block = 32 (one wave).  Each wave owns one 16-wide gate
// column slice j and computes the i/f/g/o tiles (cols j, j+512, j+1024, j+1536)
// so the nonlinearity can be applied in-register.
__global__ __launch_bounds__(32) void k_lstm(
    const int* __restrict__ ids, const float* __restrict__ h,
    const float* __restrict__ c, const float* __restrict__ embed_W,
    const float* __restrict__ w_ih, const float* __restrict__ b_ih,
    const float* __restrict__ w_hh, const float* __restrict__ b_hh,
    float* __restrict__ h_t, float* __restrict__ c_t,
    __bf16* __restrict__ comb /* (N x 2H) bf16, cols 0..H-1 */) {
  const int jt = blockIdx.x;            // 0..31  (within H)
  const int m0 = blockIdx.y * 16;       // batch tile
  const int lane = threadIdx.x;

  const float* aemb = embed_W + (size_t)ids[m0 + (lane & 15)] * E_;
  const float* ahid = h + (size_t)(m0 + (lane & 15)) * H_;

  v8f acc[4] = {v8f{}, v8f{}, v8f{}, v8f{}};   // i, f, g, o tiles

  for (int k0 = 0; k0 < E_; k0 += 32) {
    v16bf ae = load_a_frag_rowptr(aemb, k0);
    v16bf ah = load_a_frag_rowptr(ahid, k0);
#pragma unroll
    for (int q = 0; q < 4; ++q) {
      const int n0 = q * H_ + jt * 16;
      v16bf bi = load_b_fragT(w_ih, E_, n0, k0);
      v16bf bh = load_b_fragT(w_hh, H_, n0, k0);
      acc[q] = wmma_bf16(ae, bi, acc[q]);
      acc[q] = wmma_bf16(ah, bh, acc[q]);
    }
  }

  const int col = jt * 16 + (lane & 15);
  const float bi_i = b_ih[0 * H_ + col] + b_hh[0 * H_ + col];
  const float bi_f = b_ih[1 * H_ + col] + b_hh[1 * H_ + col];
  const float bi_g = b_ih[2 * H_ + col] + b_hh[2 * H_ + col];
  const float bi_o = b_ih[3 * H_ + col] + b_hh[3 * H_ + col];

#pragma unroll
  for (int r = 0; r < 8; ++r) {
    const int row = m0 + ((lane >> 4) << 3) + r;
    const float ig = acc[0][r] + bi_i;
    const float fg = acc[1][r] + bi_f;
    const float gg = acc[2][r] + bi_g;
    const float og = acc[3][r] + bi_o;
    const float cp = c[(size_t)row * H_ + col];
    const float ct = sigmoidf(fg) * cp + sigmoidf(ig) * tanhf(gg);
    const float ht = sigmoidf(og) * tanhf(ct);
    c_t[(size_t)row * H_ + col] = ct;
    h_t[(size_t)row * H_ + col] = ht;
    comb[(size_t)row * (2 * H_) + col] = (__bf16)ht;
  }
}

// hWh = h_t @ attn_Wh.T  (64 x 512).  grid = (H/16, N/16), one wave per tile.
__global__ __launch_bounds__(32) void k_hwh(
    const __bf16* __restrict__ comb,  // bf16 h_t in cols 0..H-1
    const float* __restrict__ attn_Wh, float* __restrict__ hWh) {
  const int n0 = blockIdx.x * 16;
  const int m0 = blockIdx.y * 16;
  const int lane = threadIdx.x;
  v8f acc = v8f{};
  for (int k0 = 0; k0 < H_; k0 += 32) {
    v16bf a = load_a_frag_bf(comb, 2 * H_, m0, k0);
    v16bf b = load_b_fragT(attn_Wh, H_, n0, k0);
    acc = wmma_bf16(a, b, acc);
  }
  const int col = n0 + (lane & 15);
#pragma unroll
  for (int r = 0; r < 8; ++r) {
    const int row = m0 + ((lane >> 4) << 3) + r;
    hWh[(size_t)row * H_ + col] = acc[r];
  }
}

// score partials: S = enc @ attn_Ws.T, fused tanh(hWh + S) . v reduction.
// grid = (H/16, (N*L)/16), one wave per tile; per-row dot-partials reduced
// across the 16 column lanes with shfl_xor, then atomicAdd into score[N*L].
__global__ __launch_bounds__(32) void k_attn_score(
    const float* __restrict__ enc,      // (N*L, H) row-major
    const float* __restrict__ attn_Ws,  // (H, H)
    const float* __restrict__ attn_v,   // (H)
    const float* __restrict__ hWh,      // (N, H)
    float* __restrict__ score) {        // (N*L), pre-zeroed
  const int n0 = blockIdx.x * 16;
  const int m0 = blockIdx.y * 16;       // flattened (n,l) row tile
  const int lane = threadIdx.x;
  v8f acc = v8f{};
  for (int k0 = 0; k0 < H_; k0 += 32) {
    v16bf a = load_a_frag_f32(enc, H_, m0, k0);
    v16bf b = load_b_fragT(attn_Ws, H_, n0, k0);
    acc = wmma_bf16(a, b, acc);
  }
  const int col = n0 + (lane & 15);
  const float vc = attn_v[col];
#pragma unroll
  for (int r = 0; r < 8; ++r) {
    const int row = m0 + ((lane >> 4) << 3) + r;   // flattened n*L + l
    const int n = row >> 7;                        // / L_
    float part = tanhf(acc[r] + hWh[(size_t)n * H_ + col]) * vc;
#pragma unroll
    for (int off = 1; off < 16; off <<= 1) part += __shfl_xor(part, off, 32);
    if ((lane & 15) == 0) atomicAdd(&score[row], part);
  }
}

// masked softmax over L, then context = attn_w @ enc, written as bf16 into
// combined[:, H:2H].  grid = N, block = L.
__global__ __launch_bounds__(L_) void k_softmax_ctx(
    const float* __restrict__ score, const unsigned char* __restrict__ mask,
    const float* __restrict__ enc, __bf16* __restrict__ comb) {
  const int n = blockIdx.x;
  const int t = threadIdx.x;            // l index
  __shared__ float buf[L_];
  __shared__ float w[L_];

  float s = score[n * L_ + t];
  if (mask[n * L_ + t]) s = -INFINITY;

  buf[t] = s; __syncthreads();
  for (int off = L_ / 2; off; off >>= 1) {
    if (t < off) buf[t] = fmaxf(buf[t], buf[t + off]);
    __syncthreads();
  }
  const float mx = buf[0]; __syncthreads();

  const float e = __expf(s - mx);
  buf[t] = e; __syncthreads();
  for (int off = L_ / 2; off; off >>= 1) {
    if (t < off) buf[t] += buf[t + off];
    __syncthreads();
  }
  const float inv = 1.0f / buf[0];
  w[t] = e * inv; __syncthreads();

#pragma unroll
  for (int cc = 0; cc < H_ / L_; ++cc) {
    const int col = t + cc * L_;
    float acc = 0.0f;
    for (int l = 0; l < L_; ++l)
      acc = fmaf(w[l], enc[((size_t)n * L_ + l) * H_ + col], acc);
    comb[(size_t)n * (2 * H_) + H_ + col] = (__bf16)acc;
  }
}

// logits = combined @ h2o_W.T + b.  grid = V/16, one wave per vocab tile; the
// wave keeps all four batch row-tiles in registers so h2o_W (131 MB) streams
// from HBM exactly once.
__global__ __launch_bounds__(32) void k_logits(
    const __bf16* __restrict__ comb,          // (N, 2H) bf16
    const float* __restrict__ h2o_W,          // (V, 2H)
    const float* __restrict__ h2o_b, float* __restrict__ logits) {
  const int n0 = blockIdx.x * 16;
  const int lane = threadIdx.x;
  v8f acc[N_ / 16] = {v8f{}, v8f{}, v8f{}, v8f{}};

  for (int k0 = 0; k0 < 2 * H_; k0 += 32) {
    // prefetch the weight stream two k-steps ahead
    __builtin_prefetch(h2o_W + (size_t)(n0 + (lane & 15)) * (2 * H_) + k0 + 64,
                       0, 1);
    v16bf b = load_b_fragT(h2o_W, 2 * H_, n0, k0);
#pragma unroll
    for (int mt = 0; mt < N_ / 16; ++mt) {
      v16bf a = load_a_frag_bf(comb, 2 * H_, mt * 16, k0);
      acc[mt] = wmma_bf16(a, b, acc[mt]);
    }
  }

  const int col = n0 + (lane & 15);
  const float bias = h2o_b[col];
#pragma unroll
  for (int mt = 0; mt < N_ / 16; ++mt)
#pragma unroll
    for (int r = 0; r < 8; ++r) {
      const int row = mt * 16 + ((lane >> 4) << 3) + r;
      logits[(size_t)row * V_ + col] = acc[mt][r] + bias;
    }
}

// row-wise log-softmax over V.  grid = N, block = 256.
__global__ __launch_bounds__(256) void k_logsoftmax(
    const float* __restrict__ logits, float* __restrict__ out) {
  const int n = blockIdx.x;
  const int t = threadIdx.x;
  __shared__ float buf[256];
  const float* row = logits + (size_t)n * V_;

  float mx = -INFINITY;
  for (int v = t; v < V_; v += 256) mx = fmaxf(mx, row[v]);
  buf[t] = mx; __syncthreads();
  for (int off = 128; off; off >>= 1) {
    if (t < off) buf[t] = fmaxf(buf[t], buf[t + off]);
    __syncthreads();
  }
  mx = buf[0]; __syncthreads();

  float s = 0.0f;
  for (int v = t; v < V_; v += 256) s += __expf(row[v] - mx);
  buf[t] = s; __syncthreads();
  for (int off = 128; off; off >>= 1) {
    if (t < off) buf[t] += buf[t + off];
    __syncthreads();
  }
  const float lse = mx + logf(buf[0]);

  float* orow = out + (size_t)n * V_;
  for (int v = t; v < V_; v += 256) orow[v] = row[v] - lse;
}

// ---- launcher --------------------------------------------------------------

extern "C" void kernel_launch(void* const* d_in, const int* in_sizes, int n_in,
                              void* d_out, int out_size, void* d_ws,
                              size_t ws_size, hipStream_t stream) {
  (void)in_sizes; (void)n_in; (void)out_size; (void)ws_size;
  const int*           ids     = (const int*)d_in[0];
  const float*         h       = (const float*)d_in[1];
  const float*         c       = (const float*)d_in[2];
  const float*         enc     = (const float*)d_in[3];
  const unsigned char* mask    = (const unsigned char*)d_in[4];
  const float*         embed_W = (const float*)d_in[5];
  const float*         w_ih    = (const float*)d_in[6];
  const float*         b_ih    = (const float*)d_in[7];
  const float*         w_hh    = (const float*)d_in[8];
  const float*         b_hh    = (const float*)d_in[9];
  const float*         attn_Wh = (const float*)d_in[10];
  const float*         attn_Ws = (const float*)d_in[11];
  const float*         attn_v  = (const float*)d_in[12];
  const float*         h2o_W   = (const float*)d_in[13];
  const float*         h2o_b   = (const float*)d_in[14];

  float* out  = (float*)d_out;                 // (N, V)
  float* h_t  = out + (size_t)N_ * V_;         // (N, H)
  float* c_t  = h_t + (size_t)N_ * H_;         // (N, H)

  char* ws = (char*)d_ws;
  __bf16* comb   = (__bf16*)ws;                          // N x 2H bf16 (128 KB)
  float*  hWh    = (float*)(ws + (128 << 10));           // N x H      (128 KB)
  float*  score  = (float*)(ws + (256 << 10));           // N*L        ( 32 KB)
  float*  logits = (float*)(ws + (320 << 10));           // N x V      (8.2 MB)

  k_zero<<<(N_ * L_ + 255) / 256, 256, 0, stream>>>(score, N_ * L_);
  k_lstm<<<dim3(H_ / 16, N_ / 16), 32, 0, stream>>>(
      ids, h, c, embed_W, w_ih, b_ih, w_hh, b_hh, h_t, c_t, comb);
  k_hwh<<<dim3(H_ / 16, N_ / 16), 32, 0, stream>>>(comb, attn_Wh, hWh);
  k_attn_score<<<dim3(H_ / 16, (N_ * L_) / 16), 32, 0, stream>>>(
      enc, attn_Ws, attn_v, hWh, score);
  k_softmax_ctx<<<N_, L_, 0, stream>>>(score, mask, enc, comb);
  k_logits<<<V_ / 16, 32, 0, stream>>>(comb, h2o_W, h2o_b, logits);
  k_logsoftmax<<<N_, 256, 0, stream>>>(logits, out);
}
